// MP_Layer_16406775071592
// MI455X (gfx1250) — compile-verified, matrix operand order checked by reading, not run on hardware
//
#include <hip/hip_runtime.h>

typedef _Float16 half_t;
typedef __attribute__((ext_vector_type(16))) _Float16 v16h;
typedef __attribute__((ext_vector_type(8)))  float    v8f;

// ---------------- workspace layout ----------------
// halves region (element offsets), all weight chunks stored N-major [Nout][K] f16
constexpr int HW_A_TOP = 0;        // [256][128]
constexpr int HW_A_MID = 32768;    // [256][128]
constexpr int HW_A_BOT = 65536;    // [256][128]
constexpr int HW_2A    = 98304;    // [128][256]
constexpr int HW_M_TOP = 131072;   // [256][128]
constexpr int HW_M_MID = 163840;   // [256][128]
constexpr int HW_M_BOT = 196608;   // [256][128]
constexpr int HW_2M    = 229376;   // [128][256]
constexpr int HW_1U    = 262144;   // [256][256]
constexpr int HW_2U    = 327680;   // [128][256]
constexpr int HW_NODES = 360448;   // 16*128*128 f16 nodes
constexpr int HW_H     = 622592;   // 16*128*256 f16 hidden of node MLP
constexpr int HW_TOTAL = 1146880;
constexpr size_t FBASE_BYTES = (size_t)HW_TOTAL * 2;
constexpr int F_P   = 0;           // per-batch precomputed terms: 16*4*128*256 f32
constexpr int F_AGG = 2097152;     // aggregated messages: 16*128*128 f32

// ---------------- CDNA5 async global->LDS copy (ASYNCcnt path) ----------------
// Moves 32KB (16384 halves): 256 threads x 16B x 8 offset steps. INST_OFFSET is
// added to both the LDS destination and the global source (ISA 08, GV mode).
__device__ __forceinline__ void async_copy_32k(half_t* dst, const half_t* src) {
  const int t = threadIdx.x;
  const unsigned int       lds = (unsigned int)(unsigned long long)(uintptr_t)dst + t * 16;
  const unsigned long long ga  = (unsigned long long)(uintptr_t)src + t * 16;
  asm volatile(
      "global_load_async_to_lds_b128 %0, %1, off\n\t"
      "global_load_async_to_lds_b128 %0, %1, off offset:4096\n\t"
      "global_load_async_to_lds_b128 %0, %1, off offset:8192\n\t"
      "global_load_async_to_lds_b128 %0, %1, off offset:12288\n\t"
      "global_load_async_to_lds_b128 %0, %1, off offset:16384\n\t"
      "global_load_async_to_lds_b128 %0, %1, off offset:20480\n\t"
      "global_load_async_to_lds_b128 %0, %1, off offset:24576\n\t"
      "global_load_async_to_lds_b128 %0, %1, off offset:28672"
      :: "v"(lds), "v"(ga) : "memory");
}

__device__ __forceinline__ void async_wait() {
  asm volatile("s_wait_asynccnt 0x0" ::: "memory");
}

// ---------------- WMMA helpers ----------------
// A-fragment (16x32 f16, M x K): lane holds row m=lane&15, k-base=(lane>>4)*8,
// elements {kb..kb+7, kb+16..kb+23}. B-fragments use the same layout from
// N-major (transposed) weight tiles, so B[k][n] lands with n = lane&15.
__device__ __forceinline__ v16h load_frag(const half_t* base, int ld) {
  const int lane = threadIdx.x & 31;
  const int m  = lane & 15;
  const int kb = (lane >> 4) << 3;
  union { v16h v; float4 q[2]; } u;
  u.q[0] = *(const float4*)(base + m * ld + kb);
  u.q[1] = *(const float4*)(base + m * ld + kb + 16);
  return u.v;
}

template <int NT>
__device__ __forceinline__ void zero_acc(v8f* acc) {
#pragma unroll
  for (int ct = 0; ct < NT; ++ct)
#pragma unroll
    for (int e = 0; e < 8; ++e) acc[ct][e] = 0.f;
}

// one 16-row strip (A) times NT 16-col weight tiles (W, N-major, ld=K)
template <int NT>
__device__ __forceinline__ void mma_strip(const half_t* __restrict__ A,
                                          const half_t* __restrict__ W,
                                          int K, v8f* acc) {
  for (int kt = 0; kt < (K >> 5); ++kt) {
    v16h af = load_frag(A + (kt << 5), K);
#pragma unroll
    for (int ct = 0; ct < NT; ++ct) {
      v16h bf = load_frag(W + ct * 16 * K + (kt << 5), K);
      acc[ct] = __builtin_amdgcn_wmma_f32_16x16x32_f16(
          false, af, false, bf, (short)0, acc[ct], false, false);
    }
  }
}

// ---------------- prep: f16-transpose weights, f16 nodes ----------------
__device__ void tpose(const float* __restrict__ src, half_t* __restrict__ dst,
                      int K, int N) {
  const int total = K * N;
  const int stride = gridDim.x * blockDim.x;
  for (int i = blockIdx.x * blockDim.x + threadIdx.x; i < total; i += stride) {
    const int n = i / K, k = i - n * K;
    dst[i] = (half_t)src[k * N + n];   // dst[n][k] = W[k][n]
  }
}

__global__ __launch_bounds__(256) void prep_kernel(
    const float* __restrict__ W1a, const float* __restrict__ W2a,
    const float* __restrict__ W1m, const float* __restrict__ W2m,
    const float* __restrict__ W1u, const float* __restrict__ W2u,
    const float* __restrict__ nodes, half_t* __restrict__ wsH) {
  tpose(W1a,            wsH + HW_A_TOP, 128, 256);
  tpose(W1a + 128*256,  wsH + HW_A_MID, 128, 256);
  tpose(W1a + 256*256,  wsH + HW_A_BOT, 128, 256);
  tpose(W2a,            wsH + HW_2A,    256, 128);
  tpose(W1m,            wsH + HW_M_TOP, 128, 256);
  tpose(W1m + 128*256,  wsH + HW_M_MID, 128, 256);
  tpose(W1m + 256*256,  wsH + HW_M_BOT, 128, 256);
  tpose(W2m,            wsH + HW_2M,    256, 128);
  tpose(W1u,            wsH + HW_1U,    256, 256);
  tpose(W2u,            wsH + HW_2U,    256, 128);
  const int stride = gridDim.x * blockDim.x;
  for (int i = blockIdx.x * blockDim.x + threadIdx.x; i < 16 * 128 * 128; i += stride)
    wsH[HW_NODES + i] = (half_t)nodes[i];
}

// ---------------- per-batch shared terms: Nodes_b @ {W1a_top,W1a_mid,W1m_top,W1m_mid}
__global__ __launch_bounds__(256) void batch_pre_kernel(
    const half_t* __restrict__ wsH, float* __restrict__ P) {
  __shared__ __align__(16) half_t shA[128 * 128];
  __shared__ __align__(16) half_t shW[16384];
  const int b = blockIdx.x;
  const int t = threadIdx.x, w = t >> 5, lane = t & 31;
  const int hl = lane >> 4, ln = lane & 15;
  async_copy_32k(shA, wsH + HW_NODES + b * 16384);
  const int chunkOff[4] = {HW_A_TOP, HW_A_MID, HW_M_TOP, HW_M_MID};
  for (int c = 0; c < 4; ++c) {
    float* Pc = P + (size_t)(b * 4 + c) * 32768;
    for (int seg = 0; seg < 2; ++seg) {
      __syncthreads();
      async_copy_32k(shW, wsH + chunkOff[c] + seg * 16384);
      async_wait();                 // also covers the shA copy on first pass
      __syncthreads();
      v8f acc[8];
      zero_acc<8>(acc);
      mma_strip<8>(shA + w * 16 * 128, shW, 128, acc);
#pragma unroll
      for (int ct = 0; ct < 8; ++ct) {
        const int j = seg * 128 + ct * 16 + ln;
#pragma unroll
        for (int r = 0; r < 8; ++r) {
          const int q = w * 16 + r + (hl << 3);
          Pc[q * 256 + j] = acc[ct][r];
        }
      }
    }
  }
}

// ---------------- edge blocks: one block per (b,p) ----------------
__global__ __launch_bounds__(256) void edge_kernel(
    const float* __restrict__ edges, const float* __restrict__ mask,
    const float* __restrict__ b1a, const float* __restrict__ b2a,
    const float* __restrict__ b1m, const float* __restrict__ b2m,
    const float* __restrict__ gamma_e, const float* __restrict__ beta_e,
    const float* __restrict__ mean_e, const float* __restrict__ var_e,
    const half_t* __restrict__ wsH, const float* __restrict__ P,
    float* __restrict__ agg, float* __restrict__ out) {
  __shared__ __align__(16) half_t shAct1[128 * 256];  // h1a / h1m (f16)
  __shared__ __align__(16) half_t shActE[128 * 128];  // edges, then new_edges (f16)
  __shared__ __align__(16) half_t shW[16384];         // 32KB weight slab
  __shared__ float shAdd[256];
  __shared__ float shMask[128];
  __shared__ float shSc[128], shSh[128];
  __shared__ float shAgg[128];

  const int b = blockIdx.x >> 7, p = blockIdx.x & 127;
  const int t = threadIdx.x, w = t >> 5, lane = t & 31;
  const int hl = lane >> 4, ln = lane & 15;
  const int row0 = (b << 14) + (p << 7);

  if (t < 128) {
    shMask[t] = mask[row0 + t];
    const float sc = gamma_e[t] * rsqrtf(var_e[t] + 1e-3f);
    shSc[t] = sc;
    shSh[t] = beta_e[t] - mean_e[t] * sc;
    shAgg[t] = 0.f;
  }
  {  // edges block -> f16 (conversion, stays on VALU path)
    const float* Eg = edges + (size_t)row0 * 128;
    for (int i = t; i < 16384; i += 256) shActE[i] = (half_t)Eg[i];
  }
  const float* PaTop = P + (size_t)(b * 4 + 0) * 32768;
  const float* PaMid = P + (size_t)(b * 4 + 1) * 32768 + p * 256;
  const float* PmTop = P + (size_t)(b * 4 + 2) * 32768;
  const float* PmMid = P + (size_t)(b * 4 + 3) * 32768 + p * 256;
  shAdd[t] = PaMid[t] + b1a[t];

  // GEMM1: h1a = relu(PaTop + PaMid[p] + E @ W1a_bot + b1a)   (128x256)
  for (int seg = 0; seg < 2; ++seg) {
    __syncthreads();
    async_copy_32k(shW, (half_t*)(wsH + HW_A_BOT + seg * 16384));
    async_wait();
    __syncthreads();
    v8f acc[8];
    zero_acc<8>(acc);
    mma_strip<8>(shActE + w * 16 * 128, shW, 128, acc);
#pragma unroll
    for (int ct = 0; ct < 8; ++ct) {
      const int j = seg * 128 + ct * 16 + ln;
#pragma unroll
      for (int r = 0; r < 8; ++r) {
        const int q = w * 16 + r + (hl << 3);
        const float v = acc[ct][r] + PaTop[q * 256 + j] + shAdd[j];
        shAct1[q * 256 + j] = (half_t)fmaxf(v, 0.f);
      }
    }
  }
  __syncthreads();
  if (t < 128) shAdd[t] = b2a[t];

  // GEMM2: new_edges = relu(h1a @ W2a + b2a) * mask; BN -> out  (128x128)
  for (int seg = 0; seg < 2; ++seg) {
    __syncthreads();
    async_copy_32k(shW, (half_t*)(wsH + HW_2A + seg * 16384));
    async_wait();
    __syncthreads();
    v8f acc[4];
    zero_acc<4>(acc);
    mma_strip<4>(shAct1 + w * 16 * 256, shW, 256, acc);
#pragma unroll
    for (int ct = 0; ct < 4; ++ct) {
      const int j = seg * 64 + ct * 16 + ln;
#pragma unroll
      for (int r = 0; r < 8; ++r) {
        const int q = w * 16 + r + (hl << 3);
        const float v = fmaxf(acc[ct][r] + shAdd[j], 0.f) * shMask[q];
        out[262144 + (size_t)(row0 + q) * 128 + j] = v * shSc[j] + shSh[j];
        shActE[q * 128 + j] = (half_t)v;
      }
    }
  }
  __syncthreads();
  shAdd[t] = PmMid[t] + b1m[t];

  // GEMM3: h1m = relu(PmTop + PmMid[p] + new_edges @ W1m_bot + b1m)  (128x256)
  for (int seg = 0; seg < 2; ++seg) {
    __syncthreads();
    async_copy_32k(shW, (half_t*)(wsH + HW_M_BOT + seg * 16384));
    async_wait();
    __syncthreads();
    v8f acc[8];
    zero_acc<8>(acc);
    mma_strip<8>(shActE + w * 16 * 128, shW, 128, acc);
#pragma unroll
    for (int ct = 0; ct < 8; ++ct) {
      const int j = seg * 128 + ct * 16 + ln;
#pragma unroll
      for (int r = 0; r < 8; ++r) {
        const int q = w * 16 + r + (hl << 3);
        const float v = acc[ct][r] + PmTop[q * 256 + j] + shAdd[j];
        shAct1[q * 256 + j] = (half_t)fmaxf(v, 0.f);
      }
    }
  }
  __syncthreads();
  if (t < 128) shAdd[t] = b2m[t];

  // GEMM4: messages = relu(h1m @ W2m + b2m) * mask; column-reduce over q
  for (int seg = 0; seg < 2; ++seg) {
    __syncthreads();
    async_copy_32k(shW, (half_t*)(wsH + HW_2M + seg * 16384));
    async_wait();
    __syncthreads();
    v8f acc[4];
    zero_acc<4>(acc);
    mma_strip<4>(shAct1 + w * 16 * 256, shW, 256, acc);
#pragma unroll
    for (int ct = 0; ct < 4; ++ct) {
      const int j = seg * 64 + ct * 16 + ln;
      float psum = 0.f;
#pragma unroll
      for (int r = 0; r < 8; ++r) {
        const int q = w * 16 + r + (hl << 3);
        psum += fmaxf(acc[ct][r] + shAdd[j], 0.f) * shMask[q];
      }
      atomicAdd(&shAgg[j], psum);   // ds_add_f32
    }
  }
  __syncthreads();
  if (t < 128) agg[(size_t)(b * 128 + p) * 128 + t] = shAgg[t];
}

// ---------------- node MLP layer 1 ----------------
__global__ __launch_bounds__(256) void node1_kernel(
    half_t* __restrict__ wsH, const float* __restrict__ agg,
    const float* __restrict__ b1u) {
  __shared__ __align__(16) half_t shA[128 * 256];
  __shared__ __align__(16) half_t shW[16384];
  const int b = blockIdx.x;
  const int t = threadIdx.x, w = t >> 5, lane = t & 31;
  const int hl = lane >> 4, ln = lane & 15;
  for (int i = t; i < 16384; i += 256) {
    const int q = i >> 7, j = i & 127;
    shA[q * 256 + j]       = wsH[HW_NODES + b * 16384 + i];
    shA[q * 256 + 128 + j] = (half_t)agg[(size_t)b * 16384 + i];
  }
  for (int seg = 0; seg < 4; ++seg) {
    __syncthreads();
    async_copy_32k(shW, wsH + HW_1U + seg * 16384);
    async_wait();
    __syncthreads();
    v8f acc[4];
    zero_acc<4>(acc);
    mma_strip<4>(shA + w * 16 * 256, shW, 256, acc);
#pragma unroll
    for (int ct = 0; ct < 4; ++ct) {
      const int j = seg * 64 + ct * 16 + ln;
#pragma unroll
      for (int r = 0; r < 8; ++r) {
        const int q = w * 16 + r + (hl << 3);
        wsH[HW_H + b * 32768 + q * 256 + j] = (half_t)fmaxf(acc[ct][r] + b1u[j], 0.f);
      }
    }
  }
}

// ---------------- node MLP layer 2 + BN ----------------
__global__ __launch_bounds__(256) void node2_kernel(
    const half_t* __restrict__ wsH, const float* __restrict__ b2u,
    const float* __restrict__ gamma_n, const float* __restrict__ beta_n,
    const float* __restrict__ mean_n, const float* __restrict__ var_n,
    float* __restrict__ out) {
  __shared__ __align__(16) half_t shA[128 * 256];
  __shared__ __align__(16) half_t shW[16384];
  __shared__ float shSc[128], shSh[128];
  const int b = blockIdx.x;
  const int t = threadIdx.x, w = t >> 5, lane = t & 31;
  const int hl = lane >> 4, ln = lane & 15;
  async_copy_32k(shA,         (half_t*)(wsH + HW_H + b * 32768));
  async_copy_32k(shA + 16384, (half_t*)(wsH + HW_H + b * 32768 + 16384));
  if (t < 128) {
    const float sc = gamma_n[t] * rsqrtf(var_n[t] + 1e-3f);
    shSc[t] = sc;
    shSh[t] = beta_n[t] - mean_n[t] * sc;
  }
  for (int seg = 0; seg < 2; ++seg) {
    __syncthreads();
    async_copy_32k(shW, (half_t*)(wsH + HW_2U + seg * 16384));
    async_wait();                 // also covers the shA copies on first pass
    __syncthreads();
    v8f acc[4];
    zero_acc<4>(acc);
    mma_strip<4>(shA + w * 16 * 256, shW, 256, acc);
#pragma unroll
    for (int ct = 0; ct < 4; ++ct) {
      const int j = seg * 64 + ct * 16 + ln;
#pragma unroll
      for (int r = 0; r < 8; ++r) {
        const int q = w * 16 + r + (hl << 3);
        const float v = fmaxf(acc[ct][r] + b2u[j], 0.f);
        out[(size_t)(b * 128 + q) * 128 + j] = v * shSc[j] + shSh[j];
      }
    }
  }
}

extern "C" void kernel_launch(void* const* d_in, const int* in_sizes, int n_in,
                              void* d_out, int out_size, void* d_ws, size_t ws_size,
                              hipStream_t stream) {
  const float* nodes = (const float*)d_in[0];
  const float* edges = (const float*)d_in[1];
  const float* maskp = (const float*)d_in[2];
  const float* W1a = (const float*)d_in[3];  const float* b1a = (const float*)d_in[4];
  const float* W2a = (const float*)d_in[5];  const float* b2a = (const float*)d_in[6];
  const float* W1m = (const float*)d_in[7];  const float* b1m = (const float*)d_in[8];
  const float* W2m = (const float*)d_in[9];  const float* b2m = (const float*)d_in[10];
  const float* W1u = (const float*)d_in[11]; const float* b1u = (const float*)d_in[12];
  const float* W2u = (const float*)d_in[13]; const float* b2u = (const float*)d_in[14];
  const float* gamma_n = (const float*)d_in[15]; const float* beta_n = (const float*)d_in[16];
  const float* mean_n  = (const float*)d_in[17]; const float* var_n  = (const float*)d_in[18];
  const float* gamma_e = (const float*)d_in[19]; const float* beta_e = (const float*)d_in[20];
  const float* mean_e  = (const float*)d_in[21]; const float* var_e  = (const float*)d_in[22];

  half_t* wsH = (half_t*)d_ws;
  float*  wsF = (float*)((char*)d_ws + FBASE_BYTES);
  float*  P   = wsF + F_P;
  float*  AGG = wsF + F_AGG;
  float*  out = (float*)d_out;
  (void)in_sizes; (void)n_in; (void)out_size; (void)ws_size;

  prep_kernel<<<512, 256, 0, stream>>>(W1a, W2a, W1m, W2m, W1u, W2u, nodes, wsH);
  batch_pre_kernel<<<16, 256, 0, stream>>>(wsH, P);
  edge_kernel<<<2048, 256, 0, stream>>>(edges, maskp, b1a, b2a, b1m, b2m,
                                        gamma_e, beta_e, mean_e, var_e,
                                        wsH, P, AGG, out);
  node1_kernel<<<16, 256, 0, stream>>>(wsH, AGG, b1u);
  node2_kernel<<<16, 256, 0, stream>>>(wsH, b2u, gamma_n, beta_n, mean_n, var_n, out);
}
